// _MSA1_11175504904797
// MI455X (gfx1250) — matrix operand order repaired and checked by reading, not compiled, not run on hardware
//
#include <hip/hip_runtime.h>
#include <stdint.h>

typedef _Float16 h2  __attribute__((ext_vector_type(2)));
typedef _Float16 h8  __attribute__((ext_vector_type(8)));
typedef _Float16 h16 __attribute__((ext_vector_type(16)));
typedef float    f4  __attribute__((ext_vector_type(4)));
typedef float    v8f __attribute__((ext_vector_type(8)));

#define B_    8
#define NHID  256
#define L_    2048
#define NHEAD 8
#define HDIM  64
#define D_    512
#define DO_   2048
#define NQKV  1536   // q|k|v channels concatenated

static __device__ __forceinline__ h16 h16_combine(h8 lo, h8 hi) {
  union U { struct { h8 lo, hi; } p; h16 v; } u;
  u.p.lo = lo; u.p.hi = hi;
  return u.v;
}

// ---------------------------------------------------------------------------
// Tiled transpose with f32 -> f16 conversion.
// src: [R][C] f32 row-major  ->  dst: [C][R] f16 row-major
// ---------------------------------------------------------------------------
__global__ __launch_bounds__(256)
void xpose_f32_to_f16(const float* __restrict__ src, _Float16* __restrict__ dst,
                      int R, int C, size_t srcBatch, size_t dstBatch) {
  __shared__ float tile[32][33];
  const int b = blockIdx.z;
  src += (size_t)b * srcBatch;
  dst += (size_t)b * dstBatch;
  const int x0 = blockIdx.x * 32;           // along C
  const int y0 = blockIdx.y * 32;           // along R
  const int tx = threadIdx.x, ty = threadIdx.y;
#pragma unroll
  for (int j = 0; j < 4; ++j)
    tile[ty + j * 8][tx] = src[(size_t)(y0 + ty + j * 8) * C + (x0 + tx)];
  __syncthreads();
#pragma unroll
  for (int j = 0; j < 4; ++j)
    dst[(size_t)(x0 + ty + j * 8) * R + (y0 + tx)] = (_Float16)tile[tx][ty + j * 8];
}

// ---------------------------------------------------------------------------
// WMMA GEMM:  C[M x N] = A[M x K] * Bt[N x K]^T   (f16 in, f32 accum)
// Block tile 128x128, 8 waves, wave tile 32x64.
// K staged via double-buffered LDS chunks of 64, filled by CDNA5
// GLOBAL_LOAD_ASYNC_TO_LDS_B128 (ASYNCcnt) so HBM fetch of chunk k+1
// overlaps WMMA on chunk k, with no VGPR round-trip.
// EPI==0: add bias (bq|bk|bv), store f16 [B][L][1536] via LDS tile transpose.
// EPI==1: add bo, store f32 transposed out[B][DO][L] (D-fragment is
//         M-contiguous per lane, so the transposed store is coalesced).
// ---------------------------------------------------------------------------
template <int EPI>
__global__ __launch_bounds__(256)
void gemm_wmma(const _Float16* __restrict__ A, const _Float16* __restrict__ Bt,
               int K, size_t aBatch,
               _Float16* __restrict__ qkvOut,
               const float* __restrict__ bq, const float* __restrict__ bk,
               const float* __restrict__ bv,
               float* __restrict__ outp, const float* __restrict__ bo) {
  constexpr int KC = 64;
  // +8 f16 row pad: keeps 16B row alignment and conflict-free ds_load_b128.
  __shared__ __align__(16) _Float16 Ash[2][128][72];
  __shared__ __align__(16) _Float16 Bsh[2][128][72];

  const int tid  = threadIdx.x;
  const int lane = tid & 31;
  const int wave = tid >> 5;
  const int wy   = wave >> 1;   // 0..3 along M (32 rows each)
  const int wx   = wave & 1;    // 0..1 along N (64 cols each)
  const int b    = blockIdx.z;
  const int bn0  = blockIdx.x * 128;
  const int bm0  = blockIdx.y * 128;

  const _Float16* Abase = A + (size_t)b * aBatch + (size_t)bm0 * K;
  const _Float16* Bbase = Bt + (size_t)bn0 * K;

  // Stage one 128x64 chunk of A and B into LDS buffer `buf` asynchronously.
  // 8 async b128 per wave per call -> ASYNCcnt += 8.
  auto stage = [&](int buf, int k0) {
#pragma unroll
    for (int i = 0; i < 4; ++i) {
      int v   = i * 256 + tid;      // 16B-vector id; 8 vectors per row
      int row = v >> 3;
      int col = (v & 7) * 8;
      uint32_t la = (uint32_t)(uintptr_t)&Ash[buf][row][col];
      uint64_t ga = (uint64_t)(uintptr_t)(Abase + (size_t)row * K + (k0 + col));
      asm volatile("global_load_async_to_lds_b128 %0, %1, off"
                   :: "v"(la), "v"(ga) : "memory");
      uint32_t lb = (uint32_t)(uintptr_t)&Bsh[buf][row][col];
      uint64_t gb = (uint64_t)(uintptr_t)(Bbase + (size_t)row * K + (k0 + col));
      asm volatile("global_load_async_to_lds_b128 %0, %1, off"
                   :: "v"(lb), "v"(gb) : "memory");
    }
  };

  v8f acc[2][4] = {};
  const int ml  = lane & 15;
  const int sel = lane >> 4;

  stage(0, 0);
  const int nChunks = K / KC;
  for (int c = 0; c < nChunks; ++c) {
    const int cur = c & 1;
    if (c + 1 < nChunks) {
      stage(cur ^ 1, (c + 1) * KC);
      // Retire only the previous chunk's 8 async ops; next chunk stays in flight.
      asm volatile("s_wait_asynccnt 0x8" ::: "memory");
    } else {
      asm volatile("s_wait_asynccnt 0x0" ::: "memory");
    }
    __syncthreads();   // all waves' shares of buffer `cur` are in LDS
#pragma unroll
    for (int kk = 0; kk < KC; kk += 32) {
      h16 af[2], bf[4];
#pragma unroll
      for (int mt = 0; mt < 2; ++mt) {
        int r = wy * 32 + mt * 16 + ml;
        // A 16x32 layout: lanes 0-15 K{0..7,16..23}, lanes 16-31 K{8..15,24..31}
        af[mt] = h16_combine(*(const h8*)&Ash[cur][r][kk + sel * 8],
                             *(const h8*)&Ash[cur][r][kk + 16 + sel * 8]);
      }
#pragma unroll
      for (int nt = 0; nt < 4; ++nt) {
        int r = wx * 64 + nt * 16 + ml;
        // B 32x16 layout: lanes 0-15 K{0..7,16..23}, lanes 16-31 K{8..15,24..31}
        bf[nt] = h16_combine(*(const h8*)&Bsh[cur][r][kk + sel * 8],
                             *(const h8*)&Bsh[cur][r][kk + 16 + sel * 8]);
      }
#pragma unroll
      for (int mt = 0; mt < 2; ++mt)
#pragma unroll
        for (int nt = 0; nt < 4; ++nt)
          acc[mt][nt] = __builtin_amdgcn_wmma_f32_16x16x32_f16(
              false, af[mt], false, bf[nt], (short)0, acc[mt][nt], false, false);
    }
    __syncthreads();   // buffer `cur` free before next iteration re-stages it
  }

  if constexpr (EPI == 0) {
    // Per-wave 16x16 f32 staging region inside Ash (272 floats, stride-17 pad).
    float* cs = ((float*)&Ash[0][0][0]) + wave * 272;
    __syncthreads();
#pragma unroll
    for (int mt = 0; mt < 2; ++mt) {
#pragma unroll
      for (int nt = 0; nt < 4; ++nt) {
        const int n_g = bn0 + wx * 64 + nt * 16 + ml;
        const float bias =
            (n_g < 512) ? bq[n_g] : ((n_g < 1024) ? bk[n_g - 512] : bv[n_g - 1024]);
#pragma unroll
        for (int r = 0; r < 8; ++r)
          cs[(sel * 8 + r) * 17 + ml] = acc[mt][nt][r] + bias;
        __syncthreads();
        {
          const int row  = lane >> 1;
          const int hsel = lane & 1;
          const float* s = cs + row * 17 + hsel * 8;
          h8 o;
#pragma unroll
          for (int j = 0; j < 8; ++j) o[j] = (_Float16)s[j];
          const int m_g = bm0 + wy * 32 + mt * 16 + row;
          const int n0  = bn0 + wx * 64 + nt * 16 + hsel * 8;
          *(h8*)(qkvOut + ((size_t)b * L_ + m_g) * NQKV + n0) = o;
        }
        __syncthreads();
      }
    }
  } else {
    // out[b][n][m]: lane's 8 accum elements are consecutive in m.
#pragma unroll
    for (int mt = 0; mt < 2; ++mt) {
#pragma unroll
      for (int nt = 0; nt < 4; ++nt) {
        const int   n_g  = bn0 + wx * 64 + nt * 16 + ml;
        const float bias = bo[n_g];
        const int   m0   = bm0 + wy * 32 + mt * 16 + sel * 8;
        float* dst = outp + ((size_t)b * DO_ + n_g) * L_ + m0;
        f4 lo, hi;
#pragma unroll
        for (int j = 0; j < 4; ++j) {
          lo[j] = acc[mt][nt][j] + bias;
          hi[j] = acc[mt][nt][4 + j] + bias;
        }
        *(f4*)dst       = lo;
        *(f4*)(dst + 4) = hi;
      }
    }
  }
}

// ---------------------------------------------------------------------------
// 3-tap window attention. One wave32 per (b, l, head); 2 channels per lane.
// Reference semantics: SAME zero-padding => OOB windows get score 0 (inside
// the softmax) and v = 0.
// ---------------------------------------------------------------------------
__global__ __launch_bounds__(256)
void attn3(const _Float16* __restrict__ qkv, _Float16* __restrict__ att) {
  const int lane = threadIdx.x & 31;
  const int widx = blockIdx.x * 8 + (threadIdx.x >> 5);
  const int head = widx & (NHEAD - 1);
  const int l    = (widx >> 3) & (L_ - 1);
  const int b    = widx >> 14;                 // / (NHEAD * L_)
  const size_t rowBase = ((size_t)b * L_ + l) * NQKV;
  const int c0 = head * HDIM + lane * 2;

  const h2 qp = *(const h2*)(qkv + rowBase + c0);
  const float q0 = (float)qp.x, q1 = (float)qp.y;

  float sc[3], v0[3], v1[3];
#pragma unroll
  for (int w = 0; w < 3; ++w) {
    const int lw = l + w - 1;
    const bool ok = (lw >= 0) && (lw < L_);
    float p = 0.f;
    v0[w] = 0.f; v1[w] = 0.f;
    if (ok) {
      const size_t rb = ((size_t)b * L_ + lw) * NQKV;
      const h2 kp = *(const h2*)(qkv + rb + D_ + c0);
      const h2 vp = *(const h2*)(qkv + rb + 2 * D_ + c0);
      p = q0 * (float)kp.x + q1 * (float)kp.y;
      v0[w] = (float)vp.x; v1[w] = (float)vp.y;
    }
#pragma unroll
    for (int off = 16; off > 0; off >>= 1) p += __shfl_xor(p, off, 32);
    sc[w] = ok ? p * 0.125f : 0.f;             // 1/sqrt(64)
  }
  const float mx = fmaxf(sc[0], fmaxf(sc[1], sc[2]));
  const float e0 = __expf(sc[0] - mx), e1 = __expf(sc[1] - mx), e2 = __expf(sc[2] - mx);
  const float inv = 1.f / (e0 + e1 + e2);
  const float a0 = (e0 * v0[0] + e1 * v0[1] + e2 * v0[2]) * inv;
  const float a1 = (e0 * v1[0] + e1 * v1[1] + e2 * v1[2]) * inv;
  h2 o; o.x = (_Float16)a0; o.y = (_Float16)a1;
  *(h2*)(att + ((size_t)b * L_ + l) * D_ + c0) = o;
}

// ---------------------------------------------------------------------------
extern "C" void kernel_launch(void* const* d_in, const int* in_sizes, int n_in,
                              void* d_out, int out_size, void* d_ws, size_t ws_size,
                              hipStream_t stream) {
  const float* x  = (const float*)d_in[0];
  const float* Wq = (const float*)d_in[1];
  const float* bq = (const float*)d_in[2];
  const float* Wk = (const float*)d_in[3];
  const float* bk = (const float*)d_in[4];
  const float* Wv = (const float*)d_in[5];
  const float* bv = (const float*)d_in[6];
  const float* Wo = (const float*)d_in[7];
  const float* bo = (const float*)d_in[8];
  float* out = (float*)d_out;

  // Workspace layout (f16): xt | WqkvT | WoT | qkv | att  (~78 MB total)
  _Float16* xt    = (_Float16*)d_ws;
  _Float16* WqkvT = xt    + (size_t)B_ * L_ * NHID;
  _Float16* WoT   = WqkvT + (size_t)NQKV * NHID;
  _Float16* qkv   = WoT   + (size_t)DO_ * D_;
  _Float16* att   = qkv   + (size_t)B_ * L_ * NQKV;

  const dim3 tb(32, 8);
  // x [B,256,L] -> xt [B,L,256]
  xpose_f32_to_f16<<<dim3(L_ / 32, NHID / 32, B_), tb, 0, stream>>>(
      x, xt, NHID, L_, (size_t)NHID * L_, (size_t)L_ * NHID);
  // Wq/Wk/Wv [256,512] -> rows [512,256] of WqkvT at offsets 0/512/1024
  xpose_f32_to_f16<<<dim3(D_ / 32, NHID / 32, 1), tb, 0, stream>>>(
      Wq, WqkvT, NHID, D_, 0, 0);
  xpose_f32_to_f16<<<dim3(D_ / 32, NHID / 32, 1), tb, 0, stream>>>(
      Wk, WqkvT + (size_t)D_ * NHID, NHID, D_, 0, 0);
  xpose_f32_to_f16<<<dim3(D_ / 32, NHID / 32, 1), tb, 0, stream>>>(
      Wv, WqkvT + (size_t)2 * D_ * NHID, NHID, D_, 0, 0);
  // Wo [512,2048] -> WoT [2048,512]
  xpose_f32_to_f16<<<dim3(DO_ / 32, D_ / 32, 1), tb, 0, stream>>>(
      Wo, WoT, D_, DO_, 0, 0);

  // Fused QKV projection: qkv[b][l][0:1536]
  gemm_wmma<0><<<dim3(NQKV / 128, L_ / 128, B_), 256, 0, stream>>>(
      xt, WqkvT, NHID, (size_t)L_ * NHID, qkv, bq, bk, bv, nullptr, nullptr);

  // Window attention -> att[b][l][0:512]
  attn3<<<(B_ * L_ * NHEAD) / 8, 256, 0, stream>>>(qkv, att);

  // Output projection with fused transpose: out[b][DO][L]
  gemm_wmma<1><<<dim3(DO_ / 128, L_ / 128, B_), 256, 0, stream>>>(
      att, WoT, D_, (size_t)L_ * D_, nullptr, nullptr, nullptr, nullptr, out, bo);
}
